// Head_31920196944520
// MI455X (gfx1250) — matrix operand order repaired and checked by reading, not compiled, not run on hardware
//
#include <hip/hip_runtime.h>

typedef __attribute__((ext_vector_type(16))) _Float16 v16h;
typedef __attribute__((ext_vector_type(8)))  _Float16 v8h;
typedef __attribute__((ext_vector_type(8)))  float    v8f;
typedef __attribute__((ext_vector_type(4)))  float    v4f;

#define WMMA_F16(A,B,C) __builtin_amdgcn_wmma_f32_16x16x32_f16(false,(A),false,(B),(short)0,(C),false,false)

static constexpr int Bn = 4, Tn = 2048, Cn = 1024, Hn = 64;
static constexpr float SCALE = 0.03125f;   // C^-0.5 = 1024^-0.5
static constexpr float NEG   = -1e30f;

// ---------------------------------------------------------------------------
// Kernel 0: W (C,H) f32 -> Wt (H,C) f16, for q,k,v. Coalesced writes.
// ---------------------------------------------------------------------------
__global__ void wt_transpose_kernel(const float* __restrict__ Wq,
                                    const float* __restrict__ Wk,
                                    const float* __restrict__ Wv,
                                    _Float16* __restrict__ Wt) {
  int idx = blockIdx.x * 256 + threadIdx.x;           // over 3*Hn*Cn
  int w   = idx / (Hn * Cn);
  int rem = idx % (Hn * Cn);
  int h   = rem / Cn;
  int c   = rem % Cn;
  const float* W = (w == 0) ? Wq : ((w == 1) ? Wk : Wv);
  Wt[idx] = (_Float16)W[c * Hn + h];
}

// ---------------------------------------------------------------------------
// Kernel 1: projections y = x @ W. One wave computes a 32x64 tile
// (2 M-subtiles sharing each B fragment -> 2x weight reuse; 8 WMMAs/chunk).
// 64 VGPRs of accumulators keeps peak pressure ~130 -> no spills.
// which: 0 -> qh (B,T,64), 1 -> kh (B,T,64), 2 -> vt (B,64,T) transposed.
// ---------------------------------------------------------------------------
__global__ void __launch_bounds__(32, 1)
proj_kernel(const float* __restrict__ x,
            const _Float16* __restrict__ Wt,
            _Float16* __restrict__ qh,
            _Float16* __restrict__ kh,
            _Float16* __restrict__ vt) {
  const int lane = threadIdx.x & 31;
  const int half = lane >> 4;
  const int lrow = lane & 15;
  const int r0    = blockIdx.x * 32;                  // flattened row base over B*T
  const int which = blockIdx.y;                       // 0=q,1=k,2=v
  const _Float16* Wtw = Wt + (size_t)which * (Hn * Cn);

  const float* xrow0 = x + (size_t)(r0 + lrow) * Cn;

  v8f acc[2][4];                                      // [m-tile][n-group]
#pragma unroll
  for (int t = 0; t < 2; ++t)
#pragma unroll
    for (int g = 0; g < 4; ++g) acc[t][g] = v8f{};

  for (int c = 0; c < Cn / 32; ++c) {
    const int base = c * 32 + 16 * half;
    // B fragments first: n = lrow + 16g, k = e + 16*half (contiguous 16 halves)
    const _Float16* wb = Wtw + c * 32 + 16 * half;
    v16h b0 = *(const v16h*)(wb + (size_t)(lrow +  0) * Cn);
    v16h b1 = *(const v16h*)(wb + (size_t)(lrow + 16) * Cn);
    v16h b2 = *(const v16h*)(wb + (size_t)(lrow + 32) * Cn);
    v16h b3 = *(const v16h*)(wb + (size_t)(lrow + 48) * Cn);
    // Per M-subtile: one A fragment live at a time, consumed immediately.
#pragma unroll
    for (int t = 0; t < 2; ++t) {
      const float* xr = xrow0 + (size_t)(16 * t) * Cn;
      v4f x0 = *(const v4f*)(xr + base);
      v4f x1 = *(const v4f*)(xr + base + 4);
      v4f x2 = *(const v4f*)(xr + base + 8);
      v4f x3 = *(const v4f*)(xr + base + 12);
      v16h a;
#pragma unroll
      for (int i = 0; i < 4; ++i) {
        a[i]      = (_Float16)x0[i];
        a[4 + i]  = (_Float16)x1[i];
        a[8 + i]  = (_Float16)x2[i];
        a[12 + i] = (_Float16)x3[i];
      }
      acc[t][0] = WMMA_F16(a, b0, acc[t][0]);
      acc[t][1] = WMMA_F16(a, b1, acc[t][1]);
      acc[t][2] = WMMA_F16(a, b2, acc[t][2]);
      acc[t][3] = WMMA_F16(a, b3, acc[t][3]);
    }
  }

  // C layout: VGPR r -> row m = r + 8*half, col n = lrow + 16g
  const int bb   = r0 / Tn;
  const int tbas = r0 % Tn;
#pragma unroll
  for (int t = 0; t < 2; ++t) {
#pragma unroll
    for (int r = 0; r < 8; ++r) {
      const int m = 16 * t + r + 8 * half;
#pragma unroll
      for (int g = 0; g < 4; ++g) {
        const int col = lrow + 16 * g;
        const float val = acc[t][g][r];
        if (which == 0)
          qh[(size_t)(r0 + m) * Hn + col] = (_Float16)val;
        else if (which == 1)
          kh[(size_t)(r0 + m) * Hn + col] = (_Float16)val;
        else
          vt[((size_t)bb * Hn + col) * Tn + (tbas + m)] = (_Float16)val;
      }
    }
  }
}

// ---------------------------------------------------------------------------
// One 64-key attention block: S = Q K^T, online softmax, O += P V.
// MASK=false for blocks fully below the causal diagonal.
// ---------------------------------------------------------------------------
template <bool MASK>
__device__ __forceinline__ void attn_block(
    int k0, int t0, int half, int lrow,
    const v16h& aq0, const v16h& aq1,
    const _Float16* __restrict__ khb, const _Float16* __restrict__ vtb,
    _Float16* Pld, float* mrow, float* lsum,
    v8f& o0, v8f& o1, v8f& o2, v8f& o3) {
  // ---- S = Q K^T over four 16-key subtiles ----
  v8f s[4];
#pragma unroll
  for (int j = 0; j < 4; ++j) s[j] = v8f{};
  const _Float16* kbase = khb + (size_t)(k0 + lrow) * Hn + 16 * half;
#pragma unroll
  for (int j = 0; j < 4; ++j) {
    v16h bk0 = *(const v16h*)(kbase + (size_t)(16 * j) * Hn);
    v16h bk1 = *(const v16h*)(kbase + (size_t)(16 * j) * Hn + 32);
    s[j] = WMMA_F16(aq0, bk0, s[j]);
    s[j] = WMMA_F16(aq1, bk1, s[j]);
  }

  // ---- mask + scale + online softmax (per C-layout row r) ----
#pragma unroll
  for (int r = 0; r < 8; ++r) {
    const int row = t0 + r + 8 * half;
    float v[4];
#pragma unroll
    for (int j = 0; j < 4; ++j) {
      float sv = s[j][r] * SCALE;
      if (MASK) {
        const int key = k0 + 16 * j + lrow;
        sv = (key <= row) ? sv : NEG;
      }
      v[j] = sv;
    }
    float mx = fmaxf(fmaxf(v[0], v[1]), fmaxf(v[2], v[3]));
#pragma unroll
    for (int off = 1; off < 16; off <<= 1)
      mx = fmaxf(mx, __shfl_xor(mx, off, 32));        // stays within 16-lane half
    const float mnew  = fmaxf(mrow[r], mx);
    const float alpha = __expf(mrow[r] - mnew);
    float p[4], psum = 0.f;
#pragma unroll
    for (int j = 0; j < 4; ++j) { p[j] = __expf(v[j] - mnew); psum += p[j]; }
#pragma unroll
    for (int off = 1; off < 16; off <<= 1)
      psum += __shfl_xor(psum, off, 32);
    lsum[r] = lsum[r] * alpha + psum;
    mrow[r] = mnew;
    o0[r] *= alpha; o1[r] *= alpha; o2[r] *= alpha; o3[r] *= alpha;
    const int m = r + 8 * half;
#pragma unroll
    for (int j = 0; j < 4; ++j)
      Pld[m * 64 + 16 * j + lrow] = (_Float16)p[j];   // C layout -> LDS row-major
  }

  // ---- P fragments (A layout) from LDS: contiguous b128 loads ----
  v16h ap0, ap1;
  {
    const _Float16* pr = &Pld[lrow * 64 + 16 * half];
    v8h l0 = *(const v8h*)(pr);
    v8h h0 = *(const v8h*)(pr + 8);
    v8h l1 = *(const v8h*)(pr + 32);
    v8h h1 = *(const v8h*)(pr + 40);
#pragma unroll
    for (int i = 0; i < 8; ++i) {
      ap0[i] = l0[i]; ap0[8 + i] = h0[i];
      ap1[i] = l1[i]; ap1[8 + i] = h1[i];
    }
  }

  // ---- O += P V : V^T stored (B,64,T) so B-fragments are contiguous ----
  const _Float16* vb = vtb + k0 + 16 * half;
  {
    v16h c00 = *(const v16h*)(vb + (size_t)(lrow +  0) * Tn);
    v16h c01 = *(const v16h*)(vb + (size_t)(lrow +  0) * Tn + 32);
    v16h c10 = *(const v16h*)(vb + (size_t)(lrow + 16) * Tn);
    v16h c11 = *(const v16h*)(vb + (size_t)(lrow + 16) * Tn + 32);
    o0 = WMMA_F16(ap0, c00, o0); o0 = WMMA_F16(ap1, c01, o0);
    o1 = WMMA_F16(ap0, c10, o1); o1 = WMMA_F16(ap1, c11, o1);
    v16h c20 = *(const v16h*)(vb + (size_t)(lrow + 32) * Tn);
    v16h c21 = *(const v16h*)(vb + (size_t)(lrow + 32) * Tn + 32);
    v16h c30 = *(const v16h*)(vb + (size_t)(lrow + 48) * Tn);
    v16h c31 = *(const v16h*)(vb + (size_t)(lrow + 48) * Tn + 32);
    o2 = WMMA_F16(ap0, c20, o2); o2 = WMMA_F16(ap1, c21, o2);
    o3 = WMMA_F16(ap0, c30, o3); o3 = WMMA_F16(ap1, c31, o3);
  }
}

// ---------------------------------------------------------------------------
// Kernel 2: causal flash attention. One wave per 16-query tile; 64-key blocks
// with an unmasked fast path below the diagonal.
// ---------------------------------------------------------------------------
__global__ void __launch_bounds__(32, 1)
attn_kernel(const _Float16* __restrict__ qh,
            const _Float16* __restrict__ kh,
            const _Float16* __restrict__ vt,
            float* __restrict__ out) {
  __shared__ _Float16 Pld[16 * 64];
  const int lane = threadIdx.x & 31;
  const int half = lane >> 4;
  const int lrow = lane & 15;
  const int t0   = blockIdx.x * 16;
  const int b    = blockIdx.y;
  const size_t rowbase = (size_t)b * Tn + t0;

  // Q fragments for head-dim chunks c=0,1 (A layout)
  const _Float16* qrow = qh + (rowbase + lrow) * Hn;
  v16h aq0, aq1;
  {
    v8h lo0 = *(const v8h*)(qrow + 16 * half);
    v8h hi0 = *(const v8h*)(qrow + 8 + 16 * half);
    v8h lo1 = *(const v8h*)(qrow + 32 + 16 * half);
    v8h hi1 = *(const v8h*)(qrow + 40 + 16 * half);
#pragma unroll
    for (int i = 0; i < 8; ++i) {
      aq0[i] = lo0[i]; aq0[8 + i] = hi0[i];
      aq1[i] = lo1[i]; aq1[8 + i] = hi1[i];
    }
  }

  float mrow[8], lsum[8];
  v8f o0 = {}, o1 = {}, o2 = {}, o3 = {};
#pragma unroll
  for (int r = 0; r < 8; ++r) { mrow[r] = NEG; lsum[r] = 0.f; }

  const _Float16* khb = kh + (size_t)b * Tn * Hn;
  const _Float16* vtb = vt + (size_t)b * Hn * Tn;

  const int nfull = (t0 + 1) >> 6;                    // blocks fully below diagonal
  const int nb    = (t0 + 16 + 63) >> 6;              // total causal blocks

  int kb = 0;
  for (; kb < nfull; ++kb) {
    __builtin_prefetch(khb + (size_t)(kb * 64 + 64 + lrow) * Hn, 0, 1);
    attn_block<false>(kb * 64, t0, half, lrow, aq0, aq1, khb, vtb,
                      Pld, mrow, lsum, o0, o1, o2, o3);
  }
  for (; kb < nb; ++kb) {
    attn_block<true>(kb * 64, t0, half, lrow, aq0, aq1, khb, vtb,
                     Pld, mrow, lsum, o0, o1, o2, o3);
  }

  // ---- normalize + store (f32 output) ----
#pragma unroll
  for (int r = 0; r < 8; ++r) {
    const float inv = 1.0f / lsum[r];
    const size_t orow = (rowbase + r + 8 * half) * Hn;
    out[orow + lrow]      = o0[r] * inv;
    out[orow + lrow + 16] = o1[r] * inv;
    out[orow + lrow + 32] = o2[r] * inv;
    out[orow + lrow + 48] = o3[r] * inv;
  }
}

// ---------------------------------------------------------------------------
extern "C" void kernel_launch(void* const* d_in, const int* in_sizes, int n_in,
                              void* d_out, int out_size, void* d_ws, size_t ws_size,
                              hipStream_t stream) {
  // setup_inputs order: x, Wk, Wq, Wv
  const float* x  = (const float*)d_in[0];
  const float* Wk = (const float*)d_in[1];
  const float* Wq = (const float*)d_in[2];
  const float* Wv = (const float*)d_in[3];
  float* out = (float*)d_out;

  char* ws = (char*)d_ws;
  _Float16* Wt  = (_Float16*)ws;                              // [3][64][1024] f16
  _Float16* qhp = (_Float16*)(ws + (size_t)3 * Hn * Cn * 2);  // (B,T,64) f16
  _Float16* khp = qhp + (size_t)Bn * Tn * Hn;                 // (B,T,64) f16
  _Float16* vtp = khp + (size_t)Bn * Tn * Hn;                 // (B,64,T) f16

  wt_transpose_kernel<<<dim3((3 * Hn * Cn) / 256), dim3(256), 0, stream>>>(Wq, Wk, Wv, Wt);
  proj_kernel<<<dim3((Bn * Tn) / 32, 3), dim3(32), 0, stream>>>(x, Wt, qhp, khp, vtp);
  attn_kernel<<<dim3(Tn / 16, Bn), dim3(32), 0, stream>>>(qhp, khp, vtp, out);
}
